// CausalAttnBlock_11003706213167
// MI455X (gfx1250) — compile-verified, matrix operand order checked by reading, not run on hardware
//
#include <hip/hip_runtime.h>
#include <hip/hip_bf16.h>

typedef __bf16 bf16;
typedef __attribute__((ext_vector_type(16))) __bf16 v16bf;
typedef __attribute__((ext_vector_type(8)))  float  v8f;

// ---------------------------------------------------------------------------
// Async global->LDS copy (gfx1250 GLOBAL_LOAD_ASYNC_TO_LDS_B128, ASYNCcnt).
// Builtin signature (from hipcc diagnostic): (int4 AS1*, int4 AS3*, Ii, Ii).
// ---------------------------------------------------------------------------
#if defined(__has_builtin)
#  if __has_builtin(__builtin_amdgcn_global_load_async_to_lds_b128)
#    define HAVE_ASYNC_LDS 1
#  endif
#endif
#ifndef HAVE_ASYNC_LDS
#  define HAVE_ASYNC_LDS 0
#  warning "no __builtin_amdgcn_global_load_async_to_lds_b128; using sync fallback"
#endif

typedef int v4i __attribute__((vector_size(16)));
typedef __attribute__((address_space(1))) v4i gv4i;
typedef __attribute__((address_space(3))) v4i lv4i;

__device__ __forceinline__ void async_copy16(const bf16* gsrc, bf16* ldst) {
#if HAVE_ASYNC_LDS
    __builtin_amdgcn_global_load_async_to_lds_b128(
        (gv4i*)gsrc, (lv4i*)ldst, 0, 0);
#else
    *reinterpret_cast<float4*>(ldst) = *reinterpret_cast<const float4*>(gsrc);
#endif
}
__device__ __forceinline__ void wait_async_le8() {
    asm volatile("s_wait_asynccnt 8" ::: "memory");
}
__device__ __forceinline__ void wait_async_0() {
    asm volatile("s_wait_asynccnt 0" ::: "memory");
}

__device__ __forceinline__ bf16 f2bf(float f) {
    unsigned u = __builtin_bit_cast(unsigned, f);
    u += 0x7FFFu + ((u >> 16) & 1u);              // round-to-nearest-even
    unsigned short s = (unsigned short)(u >> 16);
    return __builtin_bit_cast(bf16, s);
}

// Load a 16-element bf16 fragment as two 16B chunks (A/B WMMA lane layout).
__device__ __forceinline__ v16bf load_frag(const bf16* p0, const bf16* p1) {
    v16bf r;
    *reinterpret_cast<float4*>(&r)       = *reinterpret_cast<const float4*>(p0);
    *(reinterpret_cast<float4*>(&r) + 1) = *reinterpret_cast<const float4*>(p1);
    return r;
}

__device__ __forceinline__ v8f wmma_bf16(v16bf a, v16bf b, v8f c) {
    return __builtin_amdgcn_wmma_f32_16x16x32_bf16(
        false, a, false, b, (short)0, c, false, false);
}

// ---------------------------------------------------------------------------
// Kernel 1: QKV projection. x[8192,1024] fp32 @ Wqkv[1024,3072] fp32 + b.
// Writes Q (scaled by 0.125) / K as bf16 [B,H,T,D]; V as bf16 [B,H,D,T].
// ---------------------------------------------------------------------------
__global__ __launch_bounds__(256) void qkv_gemm_kernel(
    const float* __restrict__ x, const float* __restrict__ Wqkv,
    const float* __restrict__ bqkv,
    bf16* __restrict__ qbuf, bf16* __restrict__ kbuf, bf16* __restrict__ vbuf)
{
    __shared__ bf16 ldsA[128 * 32];   // [row][k]
    __shared__ bf16 ldsB[128 * 32];   // [n][k] (transposed)
    const int tid  = threadIdx.x;
    const int lane = tid & 31, w = tid >> 5;
    const int half = lane >> 4, ln = lane & 15;
    const int wrow = (w >> 1) * 32, wcol = (w & 1) * 64;
    const int m0 = (blockIdx.x / 24) * 128;
    const int n0 = (blockIdx.x % 24) * 128;

    v8f acc[2][4];
    for (int rs = 0; rs < 2; ++rs)
        for (int ct = 0; ct < 4; ++ct) acc[rs][ct] = {};

    for (int kk = 0; kk < 1024; kk += 32) {
        // Stage A tile 128x32 (fp32 -> bf16)
        #pragma unroll
        for (int i = 0; i < 4; ++i) {
            int j = tid + 256 * i;
            int row = j >> 3, kq = (j & 7) * 4;
            float4 f = *reinterpret_cast<const float4*>(
                x + (size_t)(m0 + row) * 1024 + kk + kq);
            bf16 b4[4] = { f2bf(f.x), f2bf(f.y), f2bf(f.z), f2bf(f.w) };
            *reinterpret_cast<float2*>(&ldsA[row * 32 + kq]) =
                *reinterpret_cast<float2*>(b4);
        }
        // Stage B tile 32x128, transposed into [n][k]
        #pragma unroll
        for (int i = 0; i < 4; ++i) {
            int j = tid + 256 * i;
            int kr = j >> 5, nq = (j & 31) * 4;
            float4 f = *reinterpret_cast<const float4*>(
                Wqkv + (size_t)(kk + kr) * 3072 + n0 + nq);
            ldsB[(nq + 0) * 32 + kr] = f2bf(f.x);
            ldsB[(nq + 1) * 32 + kr] = f2bf(f.y);
            ldsB[(nq + 2) * 32 + kr] = f2bf(f.z);
            ldsB[(nq + 3) * 32 + kr] = f2bf(f.w);
        }
        __syncthreads();

        v16bf afrag[2], bfrag[4];
        #pragma unroll
        for (int rs = 0; rs < 2; ++rs) {
            int row = wrow + rs * 16 + ln;
            afrag[rs] = load_frag(&ldsA[row * 32 + 8 * half],
                                  &ldsA[row * 32 + 16 + 8 * half]);
        }
        #pragma unroll
        for (int ct = 0; ct < 4; ++ct) {
            int n = wcol + ct * 16 + ln;
            bfrag[ct] = load_frag(&ldsB[n * 32 + 8 * half],
                                  &ldsB[n * 32 + 16 + 8 * half]);
        }
        #pragma unroll
        for (int rs = 0; rs < 2; ++rs)
            #pragma unroll
            for (int ct = 0; ct < 4; ++ct)
                acc[rs][ct] = wmma_bf16(afrag[rs], bfrag[ct], acc[rs][ct]);
        __syncthreads();
    }

    // Epilogue. Section (q/k/v) is workgroup-uniform: tile is 128 wide,
    // sections are 1024 wide, n0 is a multiple of 128.
    const int sec = n0 >> 10;  // 0=Q, 1=K, 2=V
    #pragma unroll
    for (int rs = 0; rs < 2; ++rs) {
        #pragma unroll
        for (int ct = 0; ct < 4; ++ct) {
            int n = n0 + wcol + ct * 16 + ln;
            float bias = bqkv[n];
            int c = n & 1023;
            int h = c >> 6, d = c & 63;
            if (sec == 0) {
                #pragma unroll
                for (int r = 0; r < 8; ++r) {
                    int m = m0 + wrow + rs * 16 + r + 8 * half;
                    int bb = m >> 11, t = m & 2047;
                    float v = acc[rs][ct][r] + bias;
                    qbuf[(((size_t)bb * 16 + h) * 2048 + t) * 64 + d] =
                        f2bf(v * 0.125f);
                }
            } else if (sec == 1) {
                #pragma unroll
                for (int r = 0; r < 8; ++r) {
                    int m = m0 + wrow + rs * 16 + r + 8 * half;
                    int bb = m >> 11, t = m & 2047;
                    float v = acc[rs][ct][r] + bias;
                    kbuf[(((size_t)bb * 16 + h) * 2048 + t) * 64 + d] = f2bf(v);
                }
            } else {
                #pragma unroll
                for (int r = 0; r < 8; ++r) {
                    int m = m0 + wrow + rs * 16 + r + 8 * half;
                    int bb = m >> 11, t = m & 2047;
                    float v = acc[rs][ct][r] + bias;
                    vbuf[(((size_t)bb * 16 + h) * 64 + d) * 2048 + t] = f2bf(v);
                }
            }
        }
    }
}

// ---------------------------------------------------------------------------
// Kernel 2: flash attention, causal. One workgroup = one (b,h) x 64 q-rows.
// 4 waves x 16-row strips. K/V blocks double-buffered in LDS via async
// global->LDS copies (ASYNCcnt pipeline). Writes y bf16 [B*T, 1024].
// ---------------------------------------------------------------------------
__global__ __launch_bounds__(128) void flash_attn_kernel(
    const bf16* __restrict__ qbuf, const bf16* __restrict__ kbuf,
    const bf16* __restrict__ vbuf, bf16* __restrict__ ybuf)
{
    __shared__ bf16 ldsK[2][64 * 64];     // [buf][key][d]
    __shared__ bf16 ldsV[2][64 * 64];     // [buf][d][key]
    __shared__ bf16 ldsP[4 * 16 * 64];    // per-wave P staging
    const int tid  = threadIdx.x;
    const int lane = tid & 31, w = tid >> 5;
    const int half = lane >> 4, ln = lane & 15;
    const int qb = blockIdx.x & 31;
    const int bh = blockIdx.x >> 5;
    const int b = bh >> 4, h = bh & 15;
    const int qbase = qb * 64;

    const bf16* Q = qbuf + (size_t)bh * 2048 * 64;
    const bf16* K = kbuf + (size_t)bh * 2048 * 64;
    const bf16* V = vbuf + (size_t)bh * 64 * 2048;

    const int nblocks = qb + 1;                       // causal trip count

    // Issue async loads for block 0 into buffer 0 (8 x 16B per thread).
    {
        #pragma unroll
        for (int i = 0; i < 4; ++i) {
            int j = tid + 128 * i;
            int row = j >> 3, off = (j & 7) * 8;
            async_copy16(K + (size_t)row * 64 + off,   &ldsK[0][row * 64 + off]);
            async_copy16(V + (size_t)row * 2048 + off, &ldsV[0][row * 64 + off]);
        }
    }

    // Load this wave's Q strip fragments once (A-layout, D=64 -> 2 k-chunks).
    const int qrow_lane = qbase + w * 16 + ln;
    v16bf aq[2];
    #pragma unroll
    for (int c = 0; c < 2; ++c)
        aq[c] = load_frag(Q + (size_t)qrow_lane * 64 + c * 32 + 8 * half,
                          Q + (size_t)qrow_lane * 64 + c * 32 + 16 + 8 * half);

    float mrow[8], lrow[8];
    v8f o[4];
    #pragma unroll
    for (int r = 0; r < 8; ++r) { mrow[r] = -1e30f; lrow[r] = 0.0f; }
    #pragma unroll
    for (int nt = 0; nt < 4; ++nt) o[nt] = {};

    bf16* myP = &ldsP[w * 16 * 64];

    for (int jb = 0; jb < nblocks; ++jb) {
        const int jbase = jb * 64;
        const int cur = jb & 1;

        // Prefetch next K/V block into the other buffer, then wait for the
        // current block's 8-op async group (async loads complete in order).
        if (jb + 1 < nblocks) {
            const int nbase = jbase + 64;
            bf16* dK = ldsK[cur ^ 1];
            bf16* dV = ldsV[cur ^ 1];
            #pragma unroll
            for (int i = 0; i < 4; ++i) {
                int j = tid + 128 * i;
                int row = j >> 3, off = (j & 7) * 8;
                async_copy16(K + (size_t)(nbase + row) * 64 + off, dK + row * 64 + off);
                async_copy16(V + (size_t)row * 2048 + nbase + off, dV + row * 64 + off);
            }
            wait_async_le8();
        } else {
            wait_async_0();
        }
        __syncthreads();

        const bf16* cK = ldsK[cur];
        const bf16* cV = ldsV[cur];

        // S = Q * K^T  (16 x 64 strip per wave)
        v8f s[4];
        #pragma unroll
        for (int nt = 0; nt < 4; ++nt) s[nt] = {};
        #pragma unroll
        for (int c = 0; c < 2; ++c) {
            #pragma unroll
            for (int nt = 0; nt < 4; ++nt) {
                int n = nt * 16 + ln;
                v16bf bk = load_frag(&cK[n * 64 + c * 32 + 8 * half],
                                     &cK[n * 64 + c * 32 + 16 + 8 * half]);
                s[nt] = wmma_bf16(aq[c], bk, s[nt]);
            }
        }

        // Causal mask + online softmax (row = r + 8*half, col = nt*16 + ln).
        const int qrow_c = qbase + w * 16 + 8 * half;
        #pragma unroll
        for (int r = 0; r < 8; ++r) {
            int qrow = qrow_c + r;
            float mx = -1e30f;
            #pragma unroll
            for (int nt = 0; nt < 4; ++nt) {
                int key = jbase + nt * 16 + ln;
                float sv = s[nt][r];
                if (key > qrow) sv = -1e30f;
                s[nt][r] = sv;
                mx = fmaxf(mx, sv);
            }
            #pragma unroll
            for (int off = 1; off < 16; off <<= 1)
                mx = fmaxf(mx, __shfl_xor(mx, off, 32));
            float mnew  = fmaxf(mrow[r], mx);
            float alpha = __expf(mrow[r] - mnew);
            mrow[r] = mnew;
            float rsum = 0.0f;
            #pragma unroll
            for (int nt = 0; nt < 4; ++nt) {
                float p = __expf(s[nt][r] - mnew);
                s[nt][r] = p;
                rsum += p;
            }
            #pragma unroll
            for (int off = 1; off < 16; off <<= 1)
                rsum += __shfl_xor(rsum, off, 32);
            lrow[r] = lrow[r] * alpha + rsum;
            #pragma unroll
            for (int nt = 0; nt < 4; ++nt)
                o[nt][r] *= alpha;
        }

        // Stage P (C-layout -> A-layout via per-wave LDS strip).
        #pragma unroll
        for (int nt = 0; nt < 4; ++nt)
            #pragma unroll
            for (int r = 0; r < 8; ++r)
                myP[(r + 8 * half) * 64 + nt * 16 + ln] = f2bf(s[nt][r]);

        // O += P * V
        #pragma unroll
        for (int c = 0; c < 2; ++c) {
            v16bf ap = load_frag(&myP[ln * 64 + c * 32 + 8 * half],
                                 &myP[ln * 64 + c * 32 + 16 + 8 * half]);
            #pragma unroll
            for (int nt = 0; nt < 4; ++nt) {
                int d = nt * 16 + ln;
                v16bf bv = load_frag(&cV[d * 64 + c * 32 + 8 * half],
                                     &cV[d * 64 + c * 32 + 16 + 8 * half]);
                o[nt] = wmma_bf16(ap, bv, o[nt]);
            }
        }
        __syncthreads();
    }

    // Normalize and store y bf16 at [b*2048 + qrow][h*64 + d].
    #pragma unroll
    for (int r = 0; r < 8; ++r) {
        int qrow = qbase + w * 16 + 8 * half + r;
        size_t rowoff = ((size_t)b * 2048 + qrow) * 1024 + h * 64;
        float inv = 1.0f / lrow[r];
        #pragma unroll
        for (int nt = 0; nt < 4; ++nt)
            ybuf[rowoff + nt * 16 + ln] = f2bf(o[nt][r] * inv);
    }
}

// ---------------------------------------------------------------------------
// Kernel 3: output projection. y[8192,1024] bf16 @ Wout[1024,1024] + b -> fp32.
// ---------------------------------------------------------------------------
__global__ __launch_bounds__(256) void out_gemm_kernel(
    const bf16* __restrict__ y, const float* __restrict__ Wout,
    const float* __restrict__ bout, float* __restrict__ out)
{
    __shared__ bf16 ldsA[128 * 32];
    __shared__ bf16 ldsB[128 * 32];
    const int tid  = threadIdx.x;
    const int lane = tid & 31, w = tid >> 5;
    const int half = lane >> 4, ln = lane & 15;
    const int wrow = (w >> 1) * 32, wcol = (w & 1) * 64;
    const int m0 = (blockIdx.x >> 3) * 128;
    const int n0 = (blockIdx.x & 7) * 128;

    v8f acc[2][4];
    for (int rs = 0; rs < 2; ++rs)
        for (int ct = 0; ct < 4; ++ct) acc[rs][ct] = {};

    for (int kk = 0; kk < 1024; kk += 32) {
        // A tile already bf16: 128x32 = 512 x 16B loads.
        #pragma unroll
        for (int i = 0; i < 2; ++i) {
            int j = tid + 256 * i;
            int row = j >> 2, off = (j & 3) * 8;
            *reinterpret_cast<float4*>(&ldsA[row * 32 + off]) =
                *reinterpret_cast<const float4*>(
                    y + (size_t)(m0 + row) * 1024 + kk + off);
        }
        // B tile fp32 -> bf16, transposed.
        #pragma unroll
        for (int i = 0; i < 4; ++i) {
            int j = tid + 256 * i;
            int kr = j >> 5, nq = (j & 31) * 4;
            float4 f = *reinterpret_cast<const float4*>(
                Wout + (size_t)(kk + kr) * 1024 + n0 + nq);
            ldsB[(nq + 0) * 32 + kr] = f2bf(f.x);
            ldsB[(nq + 1) * 32 + kr] = f2bf(f.y);
            ldsB[(nq + 2) * 32 + kr] = f2bf(f.z);
            ldsB[(nq + 3) * 32 + kr] = f2bf(f.w);
        }
        __syncthreads();

        v16bf afrag[2], bfrag[4];
        #pragma unroll
        for (int rs = 0; rs < 2; ++rs) {
            int row = wrow + rs * 16 + ln;
            afrag[rs] = load_frag(&ldsA[row * 32 + 8 * half],
                                  &ldsA[row * 32 + 16 + 8 * half]);
        }
        #pragma unroll
        for (int ct = 0; ct < 4; ++ct) {
            int n = wcol + ct * 16 + ln;
            bfrag[ct] = load_frag(&ldsB[n * 32 + 8 * half],
                                  &ldsB[n * 32 + 16 + 8 * half]);
        }
        #pragma unroll
        for (int rs = 0; rs < 2; ++rs)
            #pragma unroll
            for (int ct = 0; ct < 4; ++ct)
                acc[rs][ct] = wmma_bf16(afrag[rs], bfrag[ct], acc[rs][ct]);
        __syncthreads();
    }

    #pragma unroll
    for (int rs = 0; rs < 2; ++rs) {
        #pragma unroll
        for (int ct = 0; ct < 4; ++ct) {
            int n = n0 + wcol + ct * 16 + ln;
            float bias = bout[n];
            #pragma unroll
            for (int r = 0; r < 8; ++r) {
                int m = m0 + wrow + rs * 16 + r + 8 * half;
                out[(size_t)m * 1024 + n] = acc[rs][ct][r] + bias;
            }
        }
    }
}

// ---------------------------------------------------------------------------
extern "C" void kernel_launch(void* const* d_in, const int* in_sizes, int n_in,
                              void* d_out, int out_size, void* d_ws, size_t ws_size,
                              hipStream_t stream) {
    const float* x    = (const float*)d_in[0];
    const float* Wqkv = (const float*)d_in[1];
    const float* bqkv = (const float*)d_in[2];
    const float* Wout = (const float*)d_in[3];
    const float* bout = (const float*)d_in[4];
    float* out = (float*)d_out;

    const size_t SZ = (size_t)4 * 16 * 2048 * 64;   // 8,388,608 elems per buffer
    bf16* ws   = (bf16*)d_ws;
    bf16* qbuf = ws;
    bf16* kbuf = ws + SZ;
    bf16* vbuf = ws + 2 * SZ;
    bf16* ybuf = ws + 3 * SZ;

    qkv_gemm_kernel  <<<dim3(64 * 24), 256, 0, stream>>>(x, Wqkv, bqkv, qbuf, kbuf, vbuf);
    flash_attn_kernel<<<dim3(2048),    128, 0, stream>>>(qbuf, kbuf, vbuf, ybuf);
    out_gemm_kernel  <<<dim3(512),     256, 0, stream>>>(ybuf, Wout, bout, out);
}
